// Latent_40853728919869
// MI455X (gfx1250) — matrix operand order repaired and checked by reading, not compiled, not run on hardware
//
#include <hip/hip_runtime.h>
#include <hip/hip_bf16.h>
#include <math.h>

typedef _Float16 h16;
typedef __attribute__((ext_vector_type(16))) _Float16 v16h;
typedef __attribute__((ext_vector_type(8)))  _Float16 v8h;
typedef __attribute__((ext_vector_type(8)))  float    v8f;

#define NB 2048   // batch
#define LL 8      // n_latents
#define DD 64     // z_dim
#define LD 512    // L*D
#define MM 2048   // e rows per latent
#define NSPLIT 2  // n-reduction split for occupancy

#define LOG2E 1.4426950408889634f

// workspace byte offsets
#define WS_ZM16  (0u)                         // [L*N*D] f16  (2 MB)
#define WS_E16   (2u*1024u*1024u)             // [L*M*D] f16  (2 MB)
#define WS_LW    (4u*1024u*1024u)             // [L*N] f32: alpha*||z||^2*log2e
#define WS_AEN2  (WS_LW + 64u*1024u)          // [L*M] f32: alpha*||e||^2
#define WS_PSUM  (WS_AEN2 + 64u*1024u)        // [L*M][NSPLIT] f32 partial exp-sums (128 KB)

// ---------------- Kernel 1: h = z @ enc_W.T + enc_b -> z_multi (f32 out) + f16 copy
__global__ void k_enc(const float* __restrict__ z, const float* __restrict__ W,
                      const float* __restrict__ b, float* __restrict__ zmulti,
                      h16* __restrict__ zm16) {
  int gid = blockIdx.x * blockDim.x + threadIdx.x;  // 2048*512 threads
  int n = gid >> 9;
  int j = gid & 511;
  const float* zr = z + n * DD;
  const float* wr = W + j * DD;
  float acc = b[j];
#pragma unroll 8
  for (int k = 0; k < DD; ++k) acc = fmaf(zr[k], wr[k], acc);
  int l = j >> 6, d = j & 63;
  int idx = (l * NB + n) * DD + d;   // z_multi[l][n][d]
  zmulti[idx] = acc;
  zm16[idx] = (h16)acc;
}

// ---------------- Kernel 2: exponent tables + e->f16 conversion
__global__ void k_prep(const float* __restrict__ zm, const float* __restrict__ e,
                       const float* __restrict__ log_sigma,
                       float* __restrict__ lw, float* __restrict__ aen2,
                       h16* __restrict__ e16) {
  int gid = blockIdx.x * blockDim.x + threadIdx.x;  // 32768 threads
  float alpha = -0.5f * __expf(-2.0f * log_sigma[0]);
  if (gid < LL * NB) {
    const float* r = zm + (size_t)gid * DD;
    float s = 0.f;
#pragma unroll 8
    for (int k = 0; k < DD; ++k) s = fmaf(r[k], r[k], s);
    lw[gid] = alpha * s * LOG2E;            // log2-domain weight
  } else {
    int idx = gid - LL * NB;
    const float* r = e + (size_t)idx * DD;
    h16* o = e16 + (size_t)idx * DD;
    float s = 0.f;
#pragma unroll 8
    for (int k = 0; k < DD; ++k) { float v = r[k]; s = fmaf(v, v, s); o[k] = (h16)v; }
    aen2[idx] = alpha * s;                  // natural-log domain (added after log)
  }
}

// ---------------- Kernel 3: z_dec = h @ dec_W.T + dec_b (reads h out of z_multi layout)
__global__ void k_dec(const float* __restrict__ zm, const float* __restrict__ Wd,
                      const float* __restrict__ bd, float* __restrict__ zdec) {
  int tid = threadIdx.x;
  int n = blockIdx.x * 4 + (tid >> 6);
  int d = tid & 63;
  const float* wr = Wd + d * LD;
  float acc = bd[d];
#pragma unroll 4
  for (int j = 0; j < LD; ++j) {
    float hv = zm[((j >> 6) * NB + n) * DD + (j & 63)];  // h[n][j]
    acc = fmaf(hv, wr[j], acc);
  }
  zdec[n * DD + d] = acc;
}

// ---------------- Kernel 4: WMMA distance GEMM + online sum-of-exp2 over n
__global__ void __launch_bounds__(128)
k_dist(const h16* __restrict__ zm16, const h16* __restrict__ e16,
       const float* __restrict__ lwtab, const float* __restrict__ log_sigma,
       float* __restrict__ psum) {
  int s    = blockIdx.x & 1;           // n-split
  int mt   = (blockIdx.x >> 1) & 31;   // m tile (64 cols per block)
  int l    = blockIdx.x >> 6;          // 0..7
  int wave = threadIdx.x >> 5;         // 0..3
  int lane = threadIdx.x & 31;
  int half = lane >> 4;
  int r    = lane & 15;
  int m_base = mt * 64 + wave * 16;

  float k2 = __expf(-2.0f * log_sigma[0]) * LOG2E;  // (-2*alpha)*log2e

  // B operand (loop-invariant): column r = e row (m_base + r); lane<16: K 0..15, lane>=16: K 16..31
  const h16* brow = e16 + ((size_t)(l * MM + m_base + r)) * DD + half * 16;
  v16h b0, b1;
  {
    v8h x0 = *(const v8h*)(brow);
    v8h x1 = *(const v8h*)(brow + 8);
    v8h y0 = *(const v8h*)(brow + 32);
    v8h y1 = *(const v8h*)(brow + 40);
#pragma unroll
    for (int i = 0; i < 8; ++i) { b0[i]=x0[i]; b0[i+8]=x1[i]; b1[i]=y0[i]; b1[i+8]=y1[i]; }
  }

  const h16* abase   = zm16 + (size_t)l * MM * DD + half * 16;
  const float* wbase = lwtab + l * MM + half * 8;
  float sum = 0.f;

  const int nbeg = s * (NB / NSPLIT);
  const int nend = nbeg + (NB / NSPLIT);

#define LOADA(NN, A0, A1)                                              \
  {                                                                    \
    const h16* ar = abase + (size_t)((NN) + r) * DD;                   \
    v8h p0 = *(const v8h*)(ar);                                        \
    v8h p1 = *(const v8h*)(ar + 8);                                    \
    v8h q0 = *(const v8h*)(ar + 32);                                   \
    v8h q1 = *(const v8h*)(ar + 40);                                   \
    _Pragma("unroll")                                                  \
    for (int i = 0; i < 8; ++i) {                                      \
      A0[i] = p0[i]; A0[i + 8] = p1[i];                                \
      A1[i] = q0[i]; A1[i + 8] = q1[i];                                \
    }                                                                  \
  }

#define PROC(C, NN)                                                    \
  {                                                                    \
    const float4 w0 = *(const float4*)(wbase + (NN));                  \
    const float4 w1 = *(const float4*)(wbase + (NN) + 4);              \
    float wv[8] = {w0.x, w0.y, w0.z, w0.w, w1.x, w1.y, w1.z, w1.w};    \
    _Pragma("unroll")                                                  \
    for (int i = 0; i < 8; ++i)                                        \
      sum += __builtin_amdgcn_exp2f(fmaf(C[i], k2, wv[i]));            \
  }

  // 2 tiles per iteration: tile n's exp processing hides behind tile n+16's WMMAs
  for (int n = nbeg; n < nend; n += 32) {
    v16h a0, a1, a2, a3;
    LOADA(n, a0, a1);
    v8f c0 = {};
    c0 = __builtin_amdgcn_wmma_f32_16x16x32_f16(false, a0, false, b0, (short)0, c0, false, false);
    c0 = __builtin_amdgcn_wmma_f32_16x16x32_f16(false, a1, false, b1, (short)0, c0, false, false);
    LOADA(n + 16, a2, a3);
    v8f c1 = {};
    c1 = __builtin_amdgcn_wmma_f32_16x16x32_f16(false, a2, false, b0, (short)0, c1, false, false);
    c1 = __builtin_amdgcn_wmma_f32_16x16x32_f16(false, a3, false, b1, (short)0, c1, false, false);
    PROC(c0, n);
    PROC(c1, n + 16);
  }
#undef LOADA
#undef PROC

  // lanes r and r+16 hold the same column m_base+r over disjoint row halves
  sum += __shfl_xor(sum, 16, 32);
  if (half == 0) {
    int m = l * MM + m_base + r;
    psum[m * NSPLIT + s] = sum;
  }
}

// ---------------- Kernel 5: combine partials, per-column log, deterministic reduction
__global__ void k_final(const float* __restrict__ psum, const float* __restrict__ aen2,
                        const float* __restrict__ log_sigma, float* __restrict__ out_scalar) {
  __shared__ float red[256];
  float acc = 0.f;
  for (int col = threadIdx.x; col < LL * MM; col += 256) {
    float tot = psum[col * NSPLIT] + psum[col * NSPLIT + 1];
    acc += aen2[col] + __logf(tot);   // lse[l,m]
  }
  red[threadIdx.x] = acc;
  __syncthreads();
  for (int st = 128; st > 0; st >>= 1) {
    if (threadIdx.x < st) red[threadIdx.x] += red[threadIdx.x + st];
    __syncthreads();
  }
  if (threadIdx.x == 0) {
    float ls = log_sigma[0];
    out_scalar[0] = -red[0] / (float)(LL * MM)
                    + 0.5f * (float)DD * (2.0f * ls - 1.0f)
                    + __logf((float)NB);
  }
}

extern "C" void kernel_launch(void* const* d_in, const int* in_sizes, int n_in,
                              void* d_out, int out_size, void* d_ws, size_t ws_size,
                              hipStream_t stream) {
  (void)in_sizes; (void)n_in; (void)out_size; (void)ws_size;
  const float* z    = (const float*)d_in[0];
  const float* e    = (const float*)d_in[1];
  const float* encW = (const float*)d_in[2];
  const float* encb = (const float*)d_in[3];
  const float* decW = (const float*)d_in[4];
  const float* decb = (const float*)d_in[5];
  const float* lsg  = (const float*)d_in[6];

  float* out    = (float*)d_out;
  float* zdec   = out;                           // [2048*64]
  float* zmulti = out + NB * DD;                 // [8*2048*64]
  float* loss   = out + NB * DD + LL * NB * DD;  // [1]

  char* ws = (char*)d_ws;
  h16*   zm16 = (h16*)(ws + WS_ZM16);
  h16*   e16  = (h16*)(ws + WS_E16);
  float* lw   = (float*)(ws + WS_LW);
  float* aen2 = (float*)(ws + WS_AEN2);
  float* psum = (float*)(ws + WS_PSUM);

  k_enc  <<<4096, 256, 0, stream>>>(z, encW, encb, zmulti, zm16);
  k_prep <<<128,  256, 0, stream>>>(zmulti, e, lsg, lw, aen2, e16);
  k_dec  <<<512,  256, 0, stream>>>(zmulti, decW, decb, zdec);
  k_dist <<<512,  128, 0, stream>>>(zm16, e16, lw, lsg, psum);
  k_final<<<1,    256, 0, stream>>>(psum, aen2, lsg, loss);
}
